// CustomLinear_73538430042502
// MI455X (gfx1250) — 8-bit matrix operand order repaired and checked by reading, compiled, not run on hardware
//
#include <hip/hip_runtime.h>
#include <stdint.h>

typedef __attribute__((ext_vector_type(16))) int   v16i;
typedef __attribute__((ext_vector_type(8)))  float v8f;

static constexpr int M = 8192, K = 4096, N = 4096;
static constexpr int TM = 128, TN = 128, TK = 128;
static constexpr int LDSS = 144;
static constexpr int TILE_BYTES = TM * LDSS;
static constexpr int BUF_BYTES  = 2 * TILE_BYTES;
static constexpr int SMEM_BYTES = 3 * BUF_BYTES;

__device__ __forceinline__ unsigned char f32_to_e4m3_sw(float f) {
    unsigned u = __float_as_uint(f);
    unsigned char sign = (unsigned char)((u >> 24) & 0x80u);
    unsigned a = u & 0x7FFFFFFFu;
    if (a >= 0x7F800000u) return (unsigned char)(sign | 0x7Fu);
    if (a > 0x43E00000u) a = 0x43E00000u;
    int e = (int)(a >> 23) - 127;
    if (e < -10) return sign;
    unsigned m = (a & 0x7FFFFFu) | 0x800000u;
    int shift = 20 + ((e < -6) ? (-6 - e) : 0);
    unsigned q    = m >> shift;
    unsigned rem  = m & ((1u << shift) - 1u);
    unsigned half = 1u << (shift - 1);
    q += (rem > half || (rem == half && (q & 1u))) ? 1u : 0u;
    if (e < -6) return (unsigned char)(sign | q);
    if (q == 16u) { q = 8u; ++e; }
    unsigned bits = ((unsigned)(e + 7) << 3) | (q & 7u);
    if (bits > 0x7Eu) bits = 0x7Eu;
    return (unsigned char)(sign | bits);
}

__global__ void quant_e4m3_kernel(const float* __restrict__ in,
                                  unsigned* __restrict__ out, int n4) {
    int i = blockIdx.x * blockDim.x + threadIdx.x;
    if (i >= n4) return;
    float4 f = ((const float4*)in)[i];
#if __has_builtin(__builtin_amdgcn_cvt_pk_fp8_f32)
    int p = 0;
    p = __builtin_amdgcn_cvt_pk_fp8_f32(f.x, f.y, p, false);
    p = __builtin_amdgcn_cvt_pk_fp8_f32(f.z, f.w, p, true);
    out[i] = (unsigned)p;
#else
    unsigned p = (unsigned)f32_to_e4m3_sw(f.x)
               | ((unsigned)f32_to_e4m3_sw(f.y) << 8)
               | ((unsigned)f32_to_e4m3_sw(f.z) << 16)
               | ((unsigned)f32_to_e4m3_sw(f.w) << 24);
    out[i] = p;
#endif
}

__device__ __forceinline__ void async_b128(unsigned lds_off, const unsigned char* gaddr) {
    asm volatile("global_load_async_to_lds_b128 %0, %1, off"
                 :: "v"(lds_off), "v"(gaddr) : "memory");
}
__device__ __forceinline__ void wait_async0() {
    asm volatile("s_wait_asynccnt 0x0" ::: "memory");
}
__device__ __forceinline__ void wait_async8() {
    asm volatile("s_wait_asynccnt 0x8" ::: "memory");
}

__global__ __launch_bounds__(256)
void gemm_fp8_wmma(const unsigned char* __restrict__ Aq,
                   const unsigned char* __restrict__ Bq,
                   float* __restrict__ C) {
    extern __shared__ unsigned char smem[];

    const int tid   = threadIdx.x;
    const int lane  = tid & 31;
    const int wave  = tid >> 5;
    const int waveM = wave & 3;
    const int waveN = wave >> 2;
    const int l16   = lane & 15;
    const int hi    = lane >> 4;

    const int bM = blockIdx.y * TM;
    const int bN = blockIdx.x * TN;

    const unsigned ldsBaseU = (unsigned)(uintptr_t)smem;

    const unsigned char* pA[4];
    const unsigned char* pB[4];
    unsigned ldsOffA[4], ldsOffB[4];
#pragma unroll
    for (int i = 0; i < 4; ++i) {
        int c = tid + i * 256;
        int r = c >> 3, co = (c & 7) << 4;
        pA[i] = Aq + (size_t)(bM + r) * K + co;
        pB[i] = Bq + (size_t)(bN + r) * K + co;
        ldsOffA[i] = (unsigned)(r * LDSS + co);
        ldsOffB[i] = (unsigned)(TILE_BYTES + r * LDSS + co);
    }

    auto issueTiles = [&](int k0, unsigned bufOff) {
        const unsigned base = ldsBaseU + bufOff;
#pragma unroll
        for (int i = 0; i < 4; ++i) {
            async_b128(base + ldsOffA[i], pA[i] + k0);
            async_b128(base + ldsOffB[i], pB[i] + k0);
        }
    };

    v8f acc[2][4];
#pragma unroll
    for (int mt = 0; mt < 2; ++mt)
#pragma unroll
        for (int nt = 0; nt < 4; ++nt)
            acc[mt][nt] = (v8f){0.f, 0.f, 0.f, 0.f, 0.f, 0.f, 0.f, 0.f};

    const int aRow0 = waveM * 32 + l16;
    const int bRow0 = waveN * 64 + l16;

    issueTiles(0, 0u);
    issueTiles(TK, (unsigned)BUF_BYTES);

    unsigned off0 = 0u, off1 = BUF_BYTES, off2 = 2u * BUF_BYTES;

    const int nIter = K / TK;
    for (int it = 0; it < nIter; ++it) {
        if (it + 1 < nIter) wait_async8();
        else                wait_async0();
        __syncthreads();
        if (it + 2 < nIter) issueTiles((it + 2) * TK, off2);

        const unsigned char* ldsA = smem + off0;
        const unsigned char* ldsB = ldsA + TILE_BYTES;

        v16i afrag[2];
#pragma unroll
        for (int mt = 0; mt < 2; ++mt) {
            const unsigned char* p = ldsA + (aRow0 + mt * 16) * LDSS + hi * 16;
#pragma unroll
            for (int j = 0; j < 8; ++j) {
                uint2 d = *(const uint2*)(p + (j >> 1) * 32 + (j & 1) * 8);
                afrag[mt][2 * j]     = (int)d.x;
                afrag[mt][2 * j + 1] = (int)d.y;
            }
        }
        v16i bfrag[4];
#pragma unroll
        for (int nt = 0; nt < 4; ++nt) {
            const unsigned char* p = ldsB + (bRow0 + nt * 16) * LDSS + hi * 16;
#pragma unroll
            for (int q = 0; q < 4; ++q) {
                int4 d = *(const int4*)(p + q * 32);
                bfrag[nt][4 * q + 0] = d.x;
                bfrag[nt][4 * q + 1] = d.y;
                bfrag[nt][4 * q + 2] = d.z;
                bfrag[nt][4 * q + 3] = d.w;
            }
        }

#pragma unroll
        for (int mt = 0; mt < 2; ++mt)
#pragma unroll
            for (int nt = 0; nt < 4; ++nt)
                acc[mt][nt] = __builtin_amdgcn_wmma_f32_16x16x128_fp8_fp8(
                    afrag[mt], bfrag[nt], (short)0, acc[mt][nt], false, false);

        unsigned t = off0; off0 = off1; off1 = off2; off2 = t;
    }

#pragma unroll
    for (int mt = 0; mt < 2; ++mt)
#pragma unroll
        for (int nt = 0; nt < 4; ++nt) {
            const int col  = bN + waveN * 64 + nt * 16 + l16;
            const int row0 = bM + waveM * 32 + mt * 16 + hi * 8;
#pragma unroll
            for (int i = 0; i < 8; ++i)
                C[(size_t)(row0 + i) * N + col] = acc[mt][nt][i];
        }
}

extern "C" void kernel_launch(void* const* d_in, const int* in_sizes, int n_in,
                              void* d_out, int out_size, void* d_ws, size_t ws_size,
                              hipStream_t stream) {
    const float* x = (const float*)d_in[0];
    const float* W = (const float*)d_in[1];
    float* out = (float*)d_out;

    unsigned char* xq = (unsigned char*)d_ws;
    unsigned char* wq = xq + (size_t)M * K;

    const int n4x = (M * K) / 4;
    quant_e4m3_kernel<<<(n4x + 255) / 256, 256, 0, stream>>>(x, (unsigned*)xq, n4x);
    const int n4w = (N * K) / 4;
    quant_e4m3_kernel<<<(n4w + 255) / 256, 256, 0, stream>>>(W, (unsigned*)wq, n4w);

    dim3 grid(N / TN, M / TM);
    gemm_fp8_wmma<<<grid, 256, SMEM_BYTES, stream>>>(xq, wq, out);
}
